// AttentionModule_1443109012385
// MI455X (gfx1250) — matrix operand order repaired and checked by reading, not compiled, not run on hardware
//
#include <hip/hip_runtime.h>

// ---------------------------------------------------------------------------
// Criss-cross slice attention for MI455X (gfx1250, wave32, WMMA f16 16x16x32).
// - Flash-attention streaming over 64-key steps (4 S fragments per softmax
//   update), K/V tiles double-buffered in LDS via
//   global_load_async_to_lds_b128 (ASYNCcnt), one barrier per step.
// - LDS rows padded to 72 halves (36 dwords) -> conflict-free ds_load_b128.
// - Softmax row reductions as one fused asm chain: pairwise max + DPP16
//   row_ror butterfly + running-max merge (8 VALU per row, no canonicalizes).
// - Q pre-scaled by log2(e) so softmax uses raw v_exp_f32 (2^x): no per-
//   element multiply before the transcendental.
// ---------------------------------------------------------------------------

typedef __attribute__((ext_vector_type(16))) _Float16 v16h;
typedef __attribute__((ext_vector_type(8)))  _Float16 v8h;
typedef __attribute__((ext_vector_type(8)))  float    v8f;

static constexpr int kN  = 16;    // slices
static constexpr int kC  = 128;   // channels
static constexpr int kHW = 2304;  // 48*48 tokens
static constexpr int kCI = 64;    // inter channels
static constexpr int kKT = kHW / 64;  // 36 key steps of 64
static constexpr int kPad = 72;   // LDS row stride in halves (bank-conflict-free)
static constexpr float kLog2e = 1.4426950408889634f;

static __device__ __forceinline__ v16h cat8(v8h lo, v8h hi) {
  return __builtin_shufflevector(lo, hi, 0,1,2,3,4,5,6,7,8,9,10,11,12,13,14,15);
}
static __device__ __forceinline__ v16h ld16(const _Float16* p) {
  return cat8(*(const v8h*)p, *(const v8h*)(p + 8));
}

static __device__ __forceinline__ v8f wmma_f16(v16h a, v16h b, v8f c) {
  return __builtin_amdgcn_wmma_f32_16x16x32_f16(false, a, false, b,
                                                (short)0, c, false, false);
}

// mn = max(m, rowmax16(max(a,b,c,d))) across each 16-lane row, fully fused.
static __device__ __forceinline__ float rowmax4m(float a, float b, float c,
                                                 float d, float m) {
  float r, t;
  asm("v_max_num_f32_e32 %0, %2, %3\n\t"
      "v_max_num_f32_e32 %1, %4, %5\n\t"
      "v_max_num_f32_e32 %0, %0, %1\n\t"
      "v_max_num_f32_dpp %0, %0, %0 row_ror:8 row_mask:0xf bank_mask:0xf bound_ctrl:1\n\t"
      "v_max_num_f32_dpp %0, %0, %0 row_ror:4 row_mask:0xf bank_mask:0xf bound_ctrl:1\n\t"
      "v_max_num_f32_dpp %0, %0, %0 row_ror:2 row_mask:0xf bank_mask:0xf bound_ctrl:1\n\t"
      "v_max_num_f32_dpp %0, %0, %0 row_ror:1 row_mask:0xf bank_mask:0xf bound_ctrl:1\n\t"
      "v_max_num_f32_e32 %0, %0, %6"
      : "=&v"(r), "=&v"(t)
      : "v"(a), "v"(b), "v"(c), "v"(d), "v"(m));
  return r;
}
static __device__ __forceinline__ float rowsum16(float v) {
  float d;
  asm("v_add_f32_dpp %0, %1, %1 row_ror:8 row_mask:0xf bank_mask:0xf bound_ctrl:1\n\t"
      "v_add_f32_dpp %0, %0, %0 row_ror:4 row_mask:0xf bank_mask:0xf bound_ctrl:1\n\t"
      "v_add_f32_dpp %0, %0, %0 row_ror:2 row_mask:0xf bank_mask:0xf bound_ctrl:1\n\t"
      "v_add_f32_dpp %0, %0, %0 row_ror:1 row_mask:0xf bank_mask:0xf bound_ctrl:1"
      : "=&v"(d) : "v"(v));
  return d;
}

// ---------------------------------------------------------------------------
// Kernel 0: Wz (128x64 f32) -> f16
// ---------------------------------------------------------------------------
__global__ void __launch_bounds__(256) wz_to_h(const float* __restrict__ Wz,
                                               _Float16* __restrict__ Wzh) {
  int i = blockIdx.x * 256 + threadIdx.x;   // 8192 total
  Wzh[i] = (_Float16)Wz[i];
}

// ---------------------------------------------------------------------------
// Kernel 1: 1x1-conv projections -> Qh,Kh (n,t,d) f16, V transposed (n,d,t).
// Q (incl. bias) is pre-scaled by log2(e) for the exp2-domain softmax.
// ---------------------------------------------------------------------------
__global__ void __launch_bounds__(256) proj_qkv(
    const float* __restrict__ F,
    const float* __restrict__ Wq, const float* __restrict__ bq,
    const float* __restrict__ Wk, const float* __restrict__ bk,
    const float* __restrict__ Wv, const float* __restrict__ bv,
    _Float16* __restrict__ Qh, _Float16* __restrict__ Kh,
    _Float16* __restrict__ Vt) {
  __shared__ float Xs[kC][64];
  const int n  = blockIdx.y;
  const int tb = blockIdx.x * 64;
  const int tid = threadIdx.x;
#pragma unroll
  for (int i = 0; i < 32; ++i) {
    int idx = tid + i * 256;
    int c = idx >> 6, t = idx & 63;
    Xs[c][t] = F[((size_t)n * kC + c) * kHW + tb + t];
  }
  __syncthreads();
  const int t  = tid & 63;
  const int dg = tid >> 6;
  for (int d0 = 0; d0 < 16; ++d0) {
    const int d = dg * 16 + d0;
    float aq = bq[d], ak = bk[d], av = bv[d];
    const float* wq = &Wq[d * kC];
    const float* wk = &Wk[d * kC];
    const float* wv = &Wv[d * kC];
#pragma unroll 8
    for (int c = 0; c < kC; ++c) {
      float x = Xs[c][t];
      aq = fmaf(x, wq[c], aq);
      ak = fmaf(x, wk[c], ak);
      av = fmaf(x, wv[c], av);
    }
    const size_t tg = (size_t)tb + t;
    Qh[((size_t)n * kHW + tg) * kCI + d] = (_Float16)(aq * kLog2e);
    Kh[((size_t)n * kHW + tg) * kCI + d] = (_Float16)ak;
    Vt[((size_t)n * kCI + d) * kHW + tg] = (_Float16)av;
  }
}

// ---------------------------------------------------------------------------
// Kernel 2: streaming attention + output projection, fused.
// Grid (18, 16), 256 threads = 8 waves, one 16-query tile per wave.
// ---------------------------------------------------------------------------
__global__ void __launch_bounds__(256) cc_attention(
    const _Float16* __restrict__ Qh, const _Float16* __restrict__ Kh,
    const _Float16* __restrict__ Vt, const _Float16* __restrict__ Wzh,
    const float* __restrict__ bz, const float* __restrict__ F,
    float* __restrict__ out) {
  __shared__ __align__(32) _Float16 Ks[2][64 * kPad];   // [buf][key][d]
  __shared__ __align__(32) _Float16 Vs[2][64 * kPad];   // [buf][d][key]
  __shared__ __align__(32) _Float16 stage[8][16 * kPad];// per-wave P/y staging

  const int n    = blockIdx.y;
  const int tid  = threadIdx.x;
  const int w    = tid >> 5;
  const int lane = tid & 31;
  const int g    = lane >> 4;
  const int l16  = lane & 15;
  const int qbase = (blockIdx.x * 8 + w) * 16;

  // per-thread staging slots: 2 x 16B chunks each for K and V tiles
  int rowS[2], colS[2];
  unsigned ldsK[2][2], ldsV[2][2];
#pragma unroll
  for (int s = 0; s < 2; ++s) {
    int j = tid + s * 256;            // 512 chunks of 8 halves per 8-KB tile
    rowS[s] = j >> 3;
    colS[s] = (j & 7) * 8;
#pragma unroll
    for (int b = 0; b < 2; ++b) {
      ldsK[b][s] = (unsigned)(uintptr_t)&Ks[b][rowS[s] * kPad + colS[s]];
      ldsV[b][s] = (unsigned)(uintptr_t)&Vs[b][rowS[s] * kPad + colS[s]];
    }
  }

  // --- Q A-fragments (16 queries x 64 d), persistent ---
  v16h qa[2];
#pragma unroll
  for (int ks = 0; ks < 2; ++ks) {
    const _Float16* qp =
        Qh + ((size_t)n * kHW + qbase + l16) * kCI + ks * 32 + g * 16;
    qa[ks] = cat8(*(const v8h*)qp, *(const v8h*)(qp + 8));
  }

  v8f ytot[4];
#pragma unroll
  for (int i = 0; i < 4; ++i) ytot[i] = (v8f){0,0,0,0,0,0,0,0};

#pragma unroll 1
  for (int nb = 0; nb < 2; ++nb) {
    const int kn = (nb == 0) ? (n > 0 ? n - 1 : 0)
                             : (n < kN - 1 ? n + 1 : kN - 1);
    const _Float16* Kb = Kh + (size_t)kn * kHW * kCI;
    const _Float16* Vb = Vt + (size_t)kn * kCI * kHW;

    // async DMA of one 64-key K tile + V tile into LDS buffer b
    auto issue = [&](int kb, int b) {
#pragma unroll
      for (int s = 0; s < 2; ++s) {
        unsigned offK = (unsigned)(((kb + rowS[s]) * kCI + colS[s]) * 2);
        asm volatile("global_load_async_to_lds_b128 %0, %1, %2"
                     :: "v"(ldsK[b][s]), "v"(offK), "s"(Kb) : "memory");
        unsigned offV = (unsigned)((rowS[s] * kHW + kb + colS[s]) * 2);
        asm volatile("global_load_async_to_lds_b128 %0, %1, %2"
                     :: "v"(ldsV[b][s]), "v"(offV), "s"(Vb) : "memory");
      }
    };

    v8f y[4];
    float m[8], ls[8];
#pragma unroll
    for (int i = 0; i < 4; ++i) y[i] = (v8f){0,0,0,0,0,0,0,0};
#pragma unroll
    for (int r = 0; r < 8; ++r) { m[r] = -1e30f; ls[r] = 0.0f; }

    issue(0, 0);                                     // prologue: tile 0

#pragma unroll 1
    for (int kt = 0; kt < kKT; ++kt) {
      const int cur = kt & 1;
      // my async stores must have landed before publishing the tile
      asm volatile("s_wait_asynccnt 0x0" ::: "memory");
      __syncthreads();
      if (kt + 1 < kKT) issue((kt + 1) * 64, cur ^ 1);  // overlap next DMA

      const _Float16* Kc = &Ks[cur][0];
      const _Float16* Vc = &Vs[cur][0];

      // --- S = Q * K^T for 64 keys (four 16-key D fragments) ---
      v8f S[4];
#pragma unroll
      for (int f = 0; f < 4; ++f) S[f] = (v8f){0,0,0,0,0,0,0,0};
#pragma unroll
      for (int ks = 0; ks < 2; ++ks)
#pragma unroll
        for (int f = 0; f < 4; ++f)
          S[f] = wmma_f16(qa[ks],
                          ld16(&Kc[(f * 16 + l16) * kPad + ks * 32 + g * 16]),
                          S[f]);

      // --- online softmax in exp2 domain, rows r + 8g ---
      float p[4][8], cor[8];
#pragma unroll
      for (int r = 0; r < 8; ++r) {
        float mn = rowmax4m(S[0][r], S[1][r], S[2][r], S[3][r], m[r]);
        cor[r] = __builtin_amdgcn_exp2f(m[r] - mn);
        m[r] = mn;
#pragma unroll
        for (int f = 0; f < 4; ++f)
          p[f][r] = __builtin_amdgcn_exp2f(S[f][r] - mn);
        ls[r] = ls[r] * cor[r] + (p[0][r] + p[1][r]) + (p[2][r] + p[3][r]);
      }
#pragma unroll
      for (int c4 = 0; c4 < 4; ++c4)
#pragma unroll
        for (int r = 0; r < 8; ++r) y[c4][r] *= cor[r];

      // --- D-layout -> A-layout for P via per-wave LDS (same-wave in-order)
      _Float16* P = &stage[w][0];                   // 16 x 64 (stride kPad)
#pragma unroll
      for (int r = 0; r < 8; ++r) {
        int row = (r + 8 * g) * kPad;
#pragma unroll
        for (int f = 0; f < 4; ++f)
          P[row + f * 16 + l16] = (_Float16)p[f][r];
      }
      v16h pa[2];
#pragma unroll
      for (int ks = 0; ks < 2; ++ks)
        pa[ks] = cat8(*(const v8h*)&P[l16 * kPad + ks * 32 + 16 * g],
                      *(const v8h*)&P[l16 * kPad + ks * 32 + 8 + 16 * g]);

      // --- y += P * V (2 K-steps x 4 N-chunks of 16 d) ---
#pragma unroll
      for (int ks = 0; ks < 2; ++ks)
#pragma unroll
        for (int c4 = 0; c4 < 4; ++c4)
          y[c4] = wmma_f16(pa[ks],
                           ld16(&Vc[(c4 * 16 + l16) * kPad + ks * 32 + g * 16]),
                           y[c4]);
    }

    // --- normalize and accumulate both neighbors ---
#pragma unroll
    for (int r = 0; r < 8; ++r) {
      float inv = 1.0f / rowsum16(ls[r]);
#pragma unroll
      for (int c4 = 0; c4 < 4; ++c4) ytot[c4][r] += y[c4][r] * inv;
    }
  }

  // --- z = ytot * Wz^T (+2*bz) + features ---
  _Float16* Y = &stage[w][0];                       // 16 x 64 (stride kPad)
#pragma unroll
  for (int c4 = 0; c4 < 4; ++c4)
#pragma unroll
    for (int r = 0; r < 8; ++r)
      Y[(r + 8 * g) * kPad + c4 * 16 + l16] = (_Float16)ytot[c4][r];

  v16h za[2];
#pragma unroll
  for (int ks = 0; ks < 2; ++ks)
    za[ks] = cat8(*(const v8h*)&Y[l16 * kPad + ks * 32 + 16 * g],
                  *(const v8h*)&Y[l16 * kPad + ks * 32 + 8 + 16 * g]);

#pragma unroll 1
  for (int cc = 0; cc < 8; ++cc) {
    v8f z = (v8f){0,0,0,0,0,0,0,0};
#pragma unroll
    for (int ks = 0; ks < 2; ++ks) {
      v16h wb = ld16(&Wzh[(size_t)(cc * 16 + l16) * kCI + ks * 32 + g * 16]);
      z = wmma_f16(za[ks], wb, z);
    }
    const int c = cc * 16 + l16;
    const float b2 = 2.0f * bz[c];
    const size_t base = ((size_t)n * kC + c) * kHW + qbase + 8 * g;
#pragma unroll
    for (int r = 0; r < 8; ++r) out[base + r] = z[r] + b2 + F[base + r];
  }
}

// ---------------------------------------------------------------------------
extern "C" void kernel_launch(void* const* d_in, const int* in_sizes, int n_in,
                              void* d_out, int out_size, void* d_ws,
                              size_t ws_size, hipStream_t stream) {
  const float* F  = (const float*)d_in[0];
  const float* Wq = (const float*)d_in[1];
  const float* bq = (const float*)d_in[2];
  const float* Wk = (const float*)d_in[3];
  const float* bk = (const float*)d_in[4];
  const float* Wv = (const float*)d_in[5];
  const float* bv = (const float*)d_in[6];
  const float* Wz = (const float*)d_in[7];
  const float* bz = (const float*)d_in[8];
  float* out = (float*)d_out;

  _Float16* ws = (_Float16*)d_ws;
  const size_t seg = (size_t)kN * kHW * kCI;   // 2,359,296 halves
  _Float16* Qh  = ws;
  _Float16* Kh  = ws + seg;
  _Float16* Vt  = ws + 2 * seg;
  _Float16* Wzh = ws + 3 * seg;

  wz_to_h<<<dim3(32), 256, 0, stream>>>(Wz, Wzh);
  proj_qkv<<<dim3(kHW / 64, kN), 256, 0, stream>>>(F, Wq, bq, Wk, bk, Wv, bv,
                                                   Qh, Kh, Vt);
  cc_attention<<<dim3(kHW / 16 / 8, kN), 256, 0, stream>>>(Qh, Kh, Vt, Wzh, bz,
                                                           F, out);
}
